// DySepConvAtten_13408887899090
// MI455X (gfx1250) — matrix operand order repaired and checked by reading, not compiled, not run on hardware
//
#include <hip/hip_runtime.h>
#include <hip/hip_bf16.h>

typedef __bf16 bf16;
typedef __attribute__((ext_vector_type(16))) __bf16 v16bf;
typedef __attribute__((ext_vector_type(8)))  __bf16 v8bf;
typedef __attribute__((ext_vector_type(8)))  float  v8f;

#define NTOK   320         // tokens (256 value + 64 query)
#define CCH    256         // channels
#define WCOLS  323         // w_lin columns (3 + 320)
#define TT_LD  328         // padded row length of tT (bf16 elems, mult of 8, conflict-free)
#define UT_LD  264         // padded row length of Ut (bf16 elems)
#define EPSF   1e-5f

// dynamic LDS byte offsets
#define STT_BYTES (256 * TT_LD * 2)            // 167,936 : tT[cc][n]  (t transposed, bf16)
#define SUT_BYTES (256 * UT_LD * 2)            // 135,168 : Ut[cc][c]  (U transposed, bf16)
#define SR_OFF    (STT_BYTES + SUT_BYTES)      // 303,104 : r[256] f32
#define SMEM_BYTES (SR_OFF + 256 * 4)          // 304,128 total

// ---- fragment loaders -------------------------------------------------------
// A-matrix 16x32 bf16 layout: lane<16 holds K={kb..kb+7, kb+16..kb+23},
//                             lane>=16 holds K={kb+8..kb+15, kb+24..kb+31}
__device__ __forceinline__ v16bf load_afrag_lds(const bf16* rowptr, int kbase, int lane) {
  const int hoff = (lane & 16) ? 8 : 0;
  v8bf c0 = *(const v8bf*)(rowptr + kbase + hoff);
  v8bf c1 = *(const v8bf*)(rowptr + kbase + 16 + hoff);
  v16bf f;
#pragma unroll
  for (int i = 0; i < 8; ++i) { f[i] = c0[i]; f[8 + i] = c1[i]; }
  return f;
}
__device__ __forceinline__ v16bf load_afrag_f32(const float* rowptr, int kbase, int lane) {
  const int hoff = (lane & 16) ? 8 : 0;
  v16bf f;
#pragma unroll
  for (int i = 0; i < 8; ++i) {
    f[i]     = (bf16)rowptr[kbase + hoff + i];
    f[8 + i] = (bf16)rowptr[kbase + 16 + hoff + i];
  }
  return f;
}
// B-matrix 32x16 bf16 from row-major [N][K] storage: lane holds column N=base+lane%16,
// K = kbase + (lane<16 ? 0 : 8) + {0..7, 16..23}
__device__ __forceinline__ v16bf load_bfrag_lds(const bf16* rowptr, int kbase, int lane) {
  const int off = kbase + ((lane & 16) ? 8 : 0);
  v8bf c0 = *(const v8bf*)(rowptr + off);
  v8bf c1 = *(const v8bf*)(rowptr + off + 16);
  v16bf f;
#pragma unroll
  for (int i = 0; i < 8; ++i) { f[i] = c0[i]; f[8 + i] = c1[i]; }
  return f;
}
__device__ __forceinline__ v16bf load_bfrag_f32(const float* rowptr, int kbase, int lane) {
  const int off = kbase + ((lane & 16) ? 8 : 0);
  v16bf f;
#pragma unroll
  for (int i = 0; i < 8; ++i) {
    f[i]     = (bf16)rowptr[off + i];
    f[8 + i] = (bf16)rowptr[off + 16 + i];
  }
  return f;
}

__global__ __launch_bounds__(256)
void dysepconv_fused(const float* __restrict__ query,  // (B,8,8,256)
                     const float* __restrict__ value,  // (B,16,16,256)
                     const float* __restrict__ w_lin,  // (256,323)
                     const float* __restrict__ b_lin,  // (323)
                     const float* __restrict__ gamma,  // (256)
                     const float* __restrict__ beta,   // (256)
                     float* __restrict__ out)          // (B,16,16,256)
{
  extern __shared__ char smem[];
  bf16*  sTT = (bf16*)smem;                  // [256][TT_LD]  t^T
  bf16*  sUT = (bf16*)(smem + STT_BYTES);    // [256][UT_LD]  U^T
  float* sR  = (float*)(smem + SR_OFF);      // [256]

  const int b    = blockIdx.x;
  const int tid  = threadIdx.x;
  const int wave = tid >> 5;                 // 8 waves
  const int lane = tid & 31;

  const float* vb = value + (size_t)b * 256 * 256;
  const float* qb = query + (size_t)b * 64 * 256;

  // ---------------- Phase 1: dw (dynamic 3-tap weights) + conv + ReLU -> tT --
  for (int n = wave; n < NTOK; n += 8) {
    const float* xrow = (n < 256) ? (vb + n * 256) : (qb + (n - 256) * 256);
    float xv[8], xm[8], xp[8];
#pragma unroll
    for (int j = 0; j < 8; ++j) {
      const int c = lane + 32 * j;
      xv[j] = xrow[c];
      xm[j] = (c > 0)   ? xrow[c - 1] : 0.f;   // zero pad
      xp[j] = (c < 255) ? xrow[c + 1] : 0.f;
    }
    float d0 = 0.f, d1 = 0.f, d2 = 0.f;
#pragma unroll
    for (int j = 0; j < 8; ++j) {
      const int c = lane + 32 * j;
      const float* wr = w_lin + c * WCOLS;
      d0 = fmaf(xv[j], wr[0], d0);
      d1 = fmaf(xv[j], wr[1], d1);
      d2 = fmaf(xv[j], wr[2], d2);
    }
#pragma unroll
    for (int off = 16; off > 0; off >>= 1) {
      d0 += __shfl_xor(d0, off, 32);
      d1 += __shfl_xor(d1, off, 32);
      d2 += __shfl_xor(d2, off, 32);
    }
    d0 += b_lin[0]; d1 += b_lin[1]; d2 += b_lin[2];
#pragma unroll
    for (int j = 0; j < 8; ++j) {
      const int c = lane + 32 * j;
      float t = fmaf(d0, xm[j], fmaf(d1, xv[j], d2 * xp[j]));
      t = fmaxf(t, 0.f);
      sTT[c * TT_LD + n] = (bf16)t;            // transposed store
    }
  }
  __syncthreads();

  // ---------------- Phase 1.5: r[cc] = sum_n bp[n] * t[n][cc] ----------------
  if (tid < 256) {
    const bf16* trow = sTT + tid * TT_LD;
    float acc = 0.f;
    for (int n = 0; n < NTOK; ++n) acc = fmaf(b_lin[3 + n], (float)trow[n], acc);
    sR[tid] = acc;
  }
  __syncthreads();

  // ---------------- Phase 2: Ut = t^T @ Wp^T  (M=cc, K=n=320, N=c) -----------
  // A: tT rows (LDS).  B: w_lin rows (global, [c][n] contiguous).  D -> sUT row-major.
#pragma unroll 1
  for (int mi = 0; mi < 2; ++mi) {
    const int mt = wave + mi * 8;
    const bf16* arow = sTT + (mt * 16 + (lane & 15)) * TT_LD;
    v16bf A[10];
#pragma unroll
    for (int kt = 0; kt < 10; ++kt) A[kt] = load_afrag_lds(arow, kt * 32, lane);

#pragma unroll 1
    for (int nt = 0; nt < 16; ++nt) {
      const float* brow = w_lin + (nt * 16 + (lane & 15)) * WCOLS + 3;
      v8f acc = {};
#pragma unroll
      for (int kt = 0; kt < 10; ++kt) {
        v16bf Bf = load_bfrag_f32(brow, kt * 32, lane);
        acc = __builtin_amdgcn_wmma_f32_16x16x32_bf16(false, A[kt], false, Bf,
                                                      (short)0, acc, false, false);
      }
      const int rbase = mt * 16 + ((lane & 16) ? 8 : 0);
      const int col   = nt * 16 + (lane & 15);
#pragma unroll
      for (int r = 0; r < 8; ++r) sUT[(rbase + r) * UT_LD + col] = (bf16)acc[r];
    }
  }
  __syncthreads();

  // ---------------- Phase 3: out = value @ U + r, then LayerNorm -------------
  // A: value rows (global f32 -> bf16).  B: sUT rows ([cc][c] = [N][K]).
  float* ob = out + (size_t)b * 256 * 256;
#pragma unroll 1
  for (int mi = 0; mi < 2; ++mi) {
    const int mt = wave + mi * 8;
    const float* arow = vb + (mt * 16 + (lane & 15)) * 256;
    v16bf A[8];
#pragma unroll
    for (int kt = 0; kt < 8; ++kt) A[kt] = load_afrag_f32(arow, kt * 32, lane);

    v8f acc[16];
#pragma unroll
    for (int nt = 0; nt < 16; ++nt) acc[nt] = (v8f){};

#pragma unroll
    for (int nt = 0; nt < 16; ++nt) {
      const bf16* brow = sUT + (nt * 16 + (lane & 15)) * UT_LD;
#pragma unroll
      for (int kt = 0; kt < 8; ++kt) {
        v16bf Bf = load_bfrag_lds(brow, kt * 32, lane);
        acc[nt] = __builtin_amdgcn_wmma_f32_16x16x32_bf16(false, A[kt], false, Bf,
                                                          (short)0, acc[nt], false, false);
      }
      const float rv = sR[nt * 16 + (lane & 15)];
#pragma unroll
      for (int r = 0; r < 8; ++r) acc[nt][r] += rv;
    }

    // in-register LayerNorm over the 256 columns of each of this wave's 16 rows
    float sum[8], sq[8];
#pragma unroll
    for (int r = 0; r < 8; ++r) { sum[r] = 0.f; sq[r] = 0.f; }
#pragma unroll
    for (int nt = 0; nt < 16; ++nt)
#pragma unroll
      for (int r = 0; r < 8; ++r) {
        const float v = acc[nt][r];
        sum[r] += v;
        sq[r]  = fmaf(v, v, sq[r]);
      }
#pragma unroll
    for (int off = 1; off < 16; off <<= 1)
#pragma unroll
      for (int r = 0; r < 8; ++r) {
        sum[r] += __shfl_xor(sum[r], off, 32);   // lanes 0-15 / 16-31 reduce separately
        sq[r]  += __shfl_xor(sq[r],  off, 32);
      }
    float mu[8], rs[8];
#pragma unroll
    for (int r = 0; r < 8; ++r) {
      mu[r] = sum[r] * (1.f / 256.f);
      const float var = sq[r] * (1.f / 256.f) - mu[r] * mu[r];
      rs[r] = rsqrtf(var + EPSF);
    }
    const int rbase = mt * 16 + ((lane & 16) ? 8 : 0);
#pragma unroll
    for (int nt = 0; nt < 16; ++nt) {
      const int cc = nt * 16 + (lane & 15);
      const float g = gamma[cc], be = beta[cc];
#pragma unroll
      for (int r = 0; r < 8; ++r)
        ob[(rbase + r) * 256 + cc] = fmaf((acc[nt][r] - mu[r]) * rs[r], g, be);
    }
  }
}

extern "C" void kernel_launch(void* const* d_in, const int* in_sizes, int n_in,
                              void* d_out, int out_size, void* d_ws, size_t ws_size,
                              hipStream_t stream) {
  const float* query = (const float*)d_in[0];
  const float* value = (const float*)d_in[1];
  const float* w_lin = (const float*)d_in[2];
  const float* b_lin = (const float*)d_in[3];
  const float* gamma = (const float*)d_in[4];
  const float* beta  = (const float*)d_in[5];
  float* out = (float*)d_out;
  (void)in_sizes; (void)n_in; (void)out_size; (void)d_ws; (void)ws_size;

  hipFuncSetAttribute((const void*)dysepconv_fused,
                      hipFuncAttributeMaxDynamicSharedMemorySize, SMEM_BYTES);
  dysepconv_fused<<<256, 256, SMEM_BYTES, stream>>>(query, value, w_lin, b_lin,
                                                    gamma, beta, out);
}
